// CNNToLSTMCustomInterleaving_66700842107518
// MI455X (gfx1250) — matrix operand order repaired and checked by reading, not compiled, not run on hardware
//
#include <hip/hip_runtime.h>
#include <stdint.h>

// ---------------- problem constants ----------------
#define B_    16
#define T_    4096
#define E_    300
#define EP    320           // padded E (mult of 32)
#define H_    300
#define HP    320           // padded H
#define G4    1200          // 4*H
#define NPC_  300
#define NPCP  304           // padded NPC (mult of 16)
#define ROWS  (B_*T_)       // 65536
#define FC    600           // feature columns (2E)
#define FCP   608           // padded (mult of 32 for K, mult of 16 for M/N)

typedef __attribute__((ext_vector_type(16))) __bf16          bf16x16;
typedef __attribute__((ext_vector_type(8)))  float           f32x8;
typedef __attribute__((ext_vector_type(8)))  unsigned short  u16x8;
typedef __attribute__((ext_vector_type(4)))  unsigned int    u32x4;
typedef __attribute__((ext_vector_type(8)))  int             i32x8;
typedef __attribute__((ext_vector_type(4)))  int             i32x4;

static __device__ __forceinline__ unsigned short f2bf(float f){
    unsigned int u = __float_as_uint(f);
    u += 0x7FFFu + ((u >> 16) & 1u);           // round-to-nearest-even
    return (unsigned short)(u >> 16);
}
static __device__ __forceinline__ float bf2f(unsigned short h){
    return __uint_as_float(((unsigned int)h) << 16);
}
static __device__ __forceinline__ float sigmf(float x){ return 1.f / (1.f + __expf(-x)); }

static __device__ __forceinline__ f32x8 wmma_bf16(bf16x16 a, bf16x16 b, f32x8 c){
    return __builtin_amdgcn_wmma_f32_16x16x32_bf16(false, a, false, b, (short)0, c, false, false);
}

// ---------------- Tensor Data Mover (TDM) 2D tile load ----------------
// D# packing per CDNA5 ISA ch.8: group0 {count, lds_addr, global_addr[56:0], type=2},
// group1 {data_size, tensor_dim0/1, tile_dim0/1, tensor_dim0_stride}, groups 2/3 zero (2D).
// 6-arg toolchain builtin: (g0, g1, g2, g3, g4, cpol).
// dsize_code: 1 = 2-byte elements, 2 = 4-byte elements.
static __device__ __forceinline__ void tdm_load_2d(unsigned lds_off, const void* gptr,
        unsigned tile_w_elems, unsigned tile_rows,
        unsigned tensor_w_elems, unsigned tensor_rows,
        unsigned long long row_stride_elems, unsigned dsize_code){
    unsigned long long ga = (unsigned long long)(size_t)gptr;
    u32x4 g0;
    g0[0] = 1u;                                              // count=1 (valid), user mode
    g0[1] = lds_off;                                         // lds_addr (bytes)
    g0[2] = (unsigned)(ga & 0xFFFFFFFFu);                    // global_addr[31:0]
    g0[3] = (unsigned)((ga >> 32) & 0x01FFFFFFu) | (2u << 30); // global_addr[56:32], type=2
    i32x8 g1;
    g1[0] = (int)(dsize_code << 16);                         // wg_mask=0, data_size
    g1[1] = (int)((tensor_w_elems & 0xFFFFu) << 16);         // tensor_dim0[15:0]
    g1[2] = (int)(((tensor_w_elems >> 16) & 0xFFFFu) |
                  ((tensor_rows & 0xFFFFu) << 16));          // tensor_dim0[31:16], tensor_dim1[15:0]
    g1[3] = (int)(((tensor_rows >> 16) & 0xFFFFu) |
                  ((tile_w_elems & 0xFFFFu) << 16));         // tensor_dim1[31:16], tile_dim0
    g1[4] = (int)(tile_rows & 0xFFFFu);                      // tile_dim1 (tile_dim2 = 0)
    g1[5] = (int)(unsigned)(row_stride_elems & 0xFFFFFFFFull);      // dim0_stride[31:0]
    g1[6] = (int)(unsigned)((row_stride_elems >> 32) & 0xFFFFull);  // dim0_stride[47:32]
    g1[7] = 0;
    i32x4 z4 = {0, 0, 0, 0};
    i32x8 z8 = {0, 0, 0, 0, 0, 0, 0, 0};
    __builtin_amdgcn_tensor_load_to_lds(g0, g1, z4, z4, z8, 0);
}

// ---- fragment loaders (wave32, 16x16x32 bf16) ----
// A 16x32: lanes 0-15 M=lane, K elems kb..kb+15, kb=0/16 per lane half
static __device__ __forceinline__ bf16x16 load_fragA_row(const unsigned short* p, int ld, int m_base, int kc){
    int lane = threadIdx.x & 31;
    int m  = m_base + (lane & 15);
    int kb = kc + ((lane >> 4) << 4);
    const unsigned short* q = p + (size_t)m * ld + kb;
    union { bf16x16 v; u16x8 h[2]; } u;
    u.h[0] = *(const u16x8*)q;
    u.h[1] = *(const u16x8*)(q + 8);
    return u.v;
}
// A from K-major (transposed source): strided gather
static __device__ __forceinline__ bf16x16 load_fragA_strided(const unsigned short* p, int ldn, int m_base, int kc){
    int lane = threadIdx.x & 31;
    int m  = m_base + (lane & 15);
    int kb = kc + ((lane >> 4) << 4);
    union { bf16x16 v; unsigned short s[16]; } u;
#pragma unroll
    for (int i = 0; i < 8; ++i){
        u.s[i]     = p[(size_t)(kb + i)      * ldn + m];
        u.s[8 + i] = p[(size_t)(kb + 8 + i) * ldn + m];
    }
    return u.v;
}
// B 32x16: lanes 0-15 N=lane hold K=kc..kc+15, lanes 16-31 hold K=kc+16..kc+31
// packed source: row n holds column n of B contiguously in K
static __device__ __forceinline__ bf16x16 load_fragB_packed(const unsigned short* p, int ld, int n_base, int kc){
    int lane = threadIdx.x & 31;
    int n  = n_base + (lane & 15);
    int kb = kc + ((lane >> 4) << 4);
    const unsigned short* q = p + (size_t)n * ld + kb;
    union { bf16x16 v; u16x8 h[2]; } u;
    u.h[0] = *(const u16x8*)q;
    u.h[1] = *(const u16x8*)(q + 8);
    return u.v;
}
// B from natural K x N row-major: strided gather
static __device__ __forceinline__ bf16x16 load_fragB_strided(const unsigned short* p, int ldn, int n_base, int kc){
    int lane = threadIdx.x & 31;
    int n  = n_base + (lane & 15);
    int kb = kc + ((lane >> 4) << 4);
    union { bf16x16 v; unsigned short s[16]; } u;
    const unsigned short* q = p + (size_t)kb * ldn + n;
#pragma unroll
    for (int i = 0; i < 16; ++i) u.s[i] = q[(size_t)i * ldn];
    return u.v;
}

// ================= kernels =================

// embedding gather -> xc bf16 (ROWS x EP), plus m4[r] = mean over E
__global__ void k_embed_pack(const int* x, const float* emb, unsigned short* xc, float* m4){
    int r = blockIdx.x;
    int idx = x[r];
    const float* er = emb + (size_t)idx * E_;
    __shared__ float red[128];
    float s = 0.f;
    for (int e = threadIdx.x; e < EP; e += blockDim.x){
        float v = (e < E_) ? er[e] : 0.f;
        xc[(size_t)r * EP + e] = f2bf(v);
        s += v;
    }
    red[threadIdx.x] = s; __syncthreads();
    for (int o = blockDim.x >> 1; o; o >>= 1){
        if (threadIdx.x < (unsigned)o) red[threadIdx.x] += red[threadIdx.x + o];
        __syncthreads();
    }
    if (threadIdx.x == 0) m4[r] = red[0] * (1.f / E_);
}

// conv weight (O=300,I=300,K=ksz) -> wr[o][k*EP+e] bf16, padded to 320x(ksz*320)
__global__ void k_pack_conv(const float* w, unsigned short* wr, int ksz){
    size_t N = (size_t)EP * ksz * EP;
    for (size_t i = (size_t)blockIdx.x * blockDim.x + threadIdx.x; i < N; i += (size_t)gridDim.x * blockDim.x){
        int o   = (int)(i / ((size_t)ksz * EP));
        int rem = (int)(i % ((size_t)ksz * EP));
        int k = rem / EP, e = rem % EP;
        float v = (o < E_ && e < E_) ? w[((size_t)o * E_ + e) * ksz + k] : 0.f;
        wr[i] = f2bf(v);
    }
}

// generic row pack f32 (R x Kin) -> bf16 (R x Kout), zero pad K
__global__ void k_pack_rows(const float* src, unsigned short* dst, int R, int Kin, int Kout){
    size_t N = (size_t)R * Kout;
    for (size_t i = (size_t)blockIdx.x * blockDim.x + threadIdx.x; i < N; i += (size_t)gridDim.x * blockDim.x){
        int r = (int)(i / Kout), k = (int)(i % Kout);
        dst[i] = (k < Kin) ? f2bf(src[(size_t)r * Kin + k]) : 0;
    }
}

__global__ void k_zero_u16(unsigned short* d, size_t n){
    for (size_t i = (size_t)blockIdx.x * blockDim.x + threadIdx.x; i < n; i += (size_t)gridDim.x * blockDim.x) d[i] = 0;
}

// conv as implicit-im2col GEMM: M=EP(out ch), N=16*Lneed (b,j), K=ksz*EP
// output Y[(b*Lneed+j)*EP + o] = conv + bias
__global__ void k_conv_gemm(const unsigned short* xc, const unsigned short* wr, const float* bias,
                            float* Y, int ksz, int stride, int pad, int Lneed){
    long wv = ((long)blockIdx.x * blockDim.x + threadIdx.x) >> 5;
    const int Mt = EP / 16;
    long tiles = (long)Mt * Lneed;
    if (wv >= tiles) return;
    int tm = (int)(wv % Mt), tn = (int)(wv / Mt);
    int lane = threadIdx.x & 31;
    int Kp = ksz * EP;
    int n  = tn * 16 + (lane & 15);
    int bb = n / Lneed, j = n % Lneed;
    f32x8 acc = {0.f,0.f,0.f,0.f,0.f,0.f,0.f,0.f};
    for (int kc = 0; kc < Kp; kc += 32){
        bf16x16 a = load_fragA_row(wr, Kp, tm * 16, kc);
        int kb = kc + ((lane >> 4) << 4);
        int kconv = kb / EP, e0 = kb % EP;
        int u = j * stride - pad + kconv;
        union { bf16x16 v; u16x8 h[2]; } bfrag;
        if ((unsigned)u < 4096u){
            const unsigned short* q = xc + (((size_t)bb << 12) + u) * EP + e0;
            bfrag.h[0] = *(const u16x8*)q;
            bfrag.h[1] = *(const u16x8*)(q + 8);
        } else {
            u16x8 z = {0,0,0,0,0,0,0,0};
            bfrag.h[0] = z; bfrag.h[1] = z;
        }
        acc = wmma_bf16(a, bfrag.v, acc);
    }
    int m0 = tm * 16 + ((lane >> 4) << 3);
#pragma unroll
    for (int r = 0; r < 8; ++r){
        int m = m0 + r;
        float bv = (m < E_) ? bias[m] : 0.f;
        Y[(size_t)n * EP + m] = acc[r] + bv;
    }
}

// scatter conv outputs into band feature matrix F (ROWS x FCP)
__global__ void k_assemble(int band, const float* Ya, const float* Yb, float* F){
    size_t N = (size_t)ROWS * FCP;
    for (size_t i = (size_t)blockIdx.x * blockDim.x + threadIdx.x; i < N; i += (size_t)gridDim.x * blockDim.x){
        int c = (int)(i % FCP);
        size_t r = i / FCP;
        int t = (int)(r & 4095), bb = (int)(r >> 12);
        float v = 0.f;
        if (c < FC){
            int e = c >> 1; int im = c & 1;
            if (band == 0){
                if (!im){
                    if (t >= 1 && t <= 4094) v = Ya[((size_t)bb * 2047 + ((t - 1) >> 1)) * EP + e];
                } else {
                    int r4 = t & 3, j;
                    if (r4 == 1) j = (t - 1) >> 2;
                    else if (r4 == 3) j = (t - 3) >> 2;
                    else if (r4 == 0) j = (t >> 2) - 1;
                    else j = (t - 6) >> 2;
                    if (j >= 0 && j <= 1022) v = Yb[((size_t)bb * 1023 + j) * EP + e];
                }
            } else if (band == 1){
                int r6 = t % 6;
                if (!im){
                    int j = -1;
                    if (r6 == 3) j = (t + 3) / 6; else if (r6 == 5) j = (t + 1) / 6; else if (r6 == 1) j = (t - 1) / 6;
                    if (j >= 1 && j <= 682) v = Ya[((size_t)bb * 683 + j) * EP + e];
                } else {
                    int j;
                    if (r6 == 3) j = (t + 3) / 6; else if (r6 == 5) j = (t + 1) / 6; else if (r6 == 1) j = (t - 1) / 6;
                    else if (r6 == 2) j = (t - 2) / 6; else if (r6 == 4) j = (t - 4) / 6; else j = (t / 6) - 1;
                    if (j >= 0 && j <= 682) v = Yb[((size_t)bb * 683 + j) * EP + e];
                }
            } else {
                if (im){
                    int r6 = t % 6, j = -1;
                    if (r6 == 1) j = (t - 1) / 6; else if (r6 == 3) j = (t - 3) / 6; else if (r6 == 5) j = (t - 5) / 6;
                    else if (r6 == 0) j = (t / 6) - 1; else if (r6 == 2 && t >= 8) j = (t - 8) / 6;
                    if (j >= 0 && j <= 680) v = Yb[((size_t)bb * 682 + (j + 1)) * EP + e];
                }
            }
        }
        F[i] = v;
    }
}

__global__ void k_colsum(const float* F, float* colsum){
    int c = blockIdx.x;
    __shared__ float red[256];
    float s = 0.f;
    for (int r = threadIdx.x; r < ROWS; r += blockDim.x) s += F[(size_t)r * FCP + c];
    red[threadIdx.x] = s; __syncthreads();
    for (int o = blockDim.x >> 1; o; o >>= 1){
        if (threadIdx.x < (unsigned)o) red[threadIdx.x] += red[threadIdx.x + o];
        __syncthreads();
    }
    if (threadIdx.x == 0) colsum[c] = red[0];
}

__global__ void k_center_pack(const float* F, const float* colsum, unsigned short* Fc){
    size_t N = (size_t)ROWS * FCP;
    for (size_t i = (size_t)blockIdx.x * blockDim.x + threadIdx.x; i < N; i += (size_t)gridDim.x * blockDim.x){
        int c = (int)(i % FCP);
        Fc[i] = f2bf(F[i] - colsum[c] * (1.f / ROWS));
    }
}

// cov = Fc^T * Fc / (B-1) -> G (FCP x FCP)
__global__ void k_cov_gemm(const unsigned short* Fc, float* G){
    long wv = ((long)blockIdx.x * blockDim.x + threadIdx.x) >> 5;
    const int Mt = FCP / 16;
    if (wv >= (long)Mt * Mt) return;
    int tm = (int)(wv % Mt), tn = (int)(wv / Mt);
    int lane = threadIdx.x & 31;
    f32x8 acc = {0.f,0.f,0.f,0.f,0.f,0.f,0.f,0.f};
    for (int kc = 0; kc < ROWS; kc += 32){
        __builtin_prefetch(Fc + (size_t)(kc + 64) * FCP + tn * 16, 0, 1);
        bf16x16 a = load_fragA_strided(Fc, FCP, tm * 16, kc);
        bf16x16 b = load_fragB_strided(Fc, FCP, tn * 16, kc);
        acc = wmma_bf16(a, b, acc);
    }
    int n  = tn * 16 + (lane & 15);
    int m0 = tm * 16 + ((lane >> 4) << 3);
#pragma unroll
    for (int r = 0; r < 8; ++r) G[(size_t)(m0 + r) * FCP + n] = acc[r] * (1.f / (B_ - 1));
}

__global__ void k_init_identity(float* V){
    size_t N = (size_t)FCP * FCP;
    for (size_t i = (size_t)blockIdx.x * blockDim.x + threadIdx.x; i < N; i += (size_t)gridDim.x * blockDim.x)
        V[i] = (i / FCP == i % FCP) ? 1.f : 0.f;
}

// one-sided Jacobi on SPD G (eigvecs accumulate in V); single block of 1024
__global__ void k_jacobi(float* G, float* V){
    const int n = FCP;
    const int tid = threadIdx.x, wid = tid >> 5, lane = tid & 31, nw = blockDim.x >> 5;
    for (int sweep = 0; sweep < 6; ++sweep){
        for (int rd = 0; rd < n - 1; ++rd){
            for (int t = wid; t < n / 2; t += nw){
                int p, q;
                if (t == 0){ p = n - 1; q = rd % (n - 1); }
                else { p = (rd + t) % (n - 1); q = (rd - t + (n - 1)) % (n - 1); }
                float app = 0.f, aqq = 0.f, apq = 0.f;
                for (int i = lane; i < n; i += 32){
                    float gp = G[(size_t)i * n + p], gq = G[(size_t)i * n + q];
                    app += gp * gp; aqq += gq * gq; apq += gp * gq;
                }
                for (int off = 16; off; off >>= 1){
                    app += __shfl_down(app, off);
                    aqq += __shfl_down(aqq, off);
                    apq += __shfl_down(apq, off);
                }
                app = __shfl(app, 0); aqq = __shfl(aqq, 0); apq = __shfl(apq, 0);
                if (apq * apq > 1e-24f * app * aqq + 1e-38f){
                    float tau = (aqq - app) / (2.f * apq);
                    float tt  = (tau >= 0.f ? 1.f : -1.f) / (fabsf(tau) + sqrtf(1.f + tau * tau));
                    float cth = 1.f / sqrtf(1.f + tt * tt);
                    float sth = tt * cth;
                    for (int i = lane; i < n; i += 32){
                        size_t ip = (size_t)i * n + p, iq = (size_t)i * n + q;
                        float gp = G[ip], gq = G[iq];
                        G[ip] = cth * gp - sth * gq;
                        G[iq] = sth * gp + cth * gq;
                        float vp = V[ip], vq = V[iq];
                        V[ip] = cth * vp - sth * vq;
                        V[iq] = sth * vp + cth * vq;
                    }
                }
            }
            __syncthreads();
        }
    }
}

// rank eigenvalues (column norms of G), pack top-300 eigenvectors as B-operand rows
__global__ void k_eig_pack(const float* G, const float* V, unsigned short* top){
    __shared__ float lam[FCP];
    int tid = threadIdx.x;
    for (int c = tid; c < FCP; c += blockDim.x){
        float s = 0.f;
        for (int i = 0; i < FCP; ++i){ float g = G[(size_t)i * FCP + c]; s += g * g; }
        lam[c] = s;
    }
    for (size_t i = tid; i < (size_t)NPCP * FCP; i += blockDim.x) top[i] = 0;
    __syncthreads();
    for (int c = tid; c < FCP; c += blockDim.x){
        float l = lam[c]; int rank = 0;
        for (int c2 = 0; c2 < FCP; ++c2){
            float l2 = lam[c2];
            rank += (l2 > l) || (l2 == l && c2 < c);
        }
        if (rank < NPC_)
            for (int k = 0; k < FCP; ++k)
                top[(size_t)rank * FCP + k] = f2bf(V[(size_t)k * FCP + c]);
    }
}

// projection: xseq = Fc (ROWS x FCP) @ top^T-pack -> bf16 (ROWS x HP)
// TDM stages a 128-row A-block of Fc into LDS; 8 waves sweep the 19 N-tiles.
__global__ void k_proj_gemm(const unsigned short* Fc, const unsigned short* top, unsigned short* xseq){
    extern __shared__ char smem[];
    unsigned short* As = (unsigned short*)smem;           // 128 x FCP bf16 = 152KB
    int wid = threadIdx.x >> 5, lane = threadIdx.x & 31;
    int mblk = blockIdx.x;                                // ROWS/128 blocks
    if (wid == 0){
        tdm_load_2d((unsigned)(size_t)(void*)As, Fc + (size_t)mblk * 128 * FCP,
                    FCP, 128, FCP, ROWS, FCP, 1);
        __builtin_amdgcn_s_wait_tensorcnt(0);
    }
    __syncthreads();
    for (int tn = wid; tn < NPCP / 16; tn += 8){
        for (int tm = 0; tm < 8; ++tm){
            f32x8 acc = {0.f,0.f,0.f,0.f,0.f,0.f,0.f,0.f};
            for (int kc = 0; kc < FCP; kc += 32){
                bf16x16 a = load_fragA_row(As, FCP, tm * 16, kc);      // from LDS
                bf16x16 b = load_fragB_packed(top, FCP, tn * 16, kc);
                acc = wmma_bf16(a, b, acc);
            }
            int n  = tn * 16 + (lane & 15);
            int m0 = mblk * 128 + tm * 16 + ((lane >> 4) << 3);
#pragma unroll
            for (int r = 0; r < 8; ++r) xseq[(size_t)(m0 + r) * HP + n] = f2bf(acc[r]);
        }
    }
}

// pre = xseq @ Wih^T + bih + bhh  -> f32 (ROWS x 1200)
// TDM stages a 128-row A-block of xseq (80KB) into LDS; 8 waves sweep 75 N-tiles.
__global__ void k_pre_gemm(const unsigned short* xseq, const unsigned short* wih,
                           const float* bih, const float* bhh, float* pre){
    extern __shared__ char smem[];
    unsigned short* As = (unsigned short*)smem;           // 128 x HP bf16 = 80KB
    int wid = threadIdx.x >> 5, lane = threadIdx.x & 31;
    int mblk = blockIdx.x;                                // ROWS/128 blocks
    if (wid == 0){
        tdm_load_2d((unsigned)(size_t)(void*)As, xseq + (size_t)mblk * 128 * HP,
                    HP, 128, HP, ROWS, HP, 1);
        __builtin_amdgcn_s_wait_tensorcnt(0);
    }
    __syncthreads();
    for (int tn = wid; tn < G4 / 16; tn += 8){
        int n  = tn * 16 + (lane & 15);
        float bv = bih[n] + bhh[n];
        for (int tm = 0; tm < 8; ++tm){
            f32x8 acc = {0.f,0.f,0.f,0.f,0.f,0.f,0.f,0.f};
            for (int kc = 0; kc < HP; kc += 32){
                bf16x16 a = load_fragA_row(As, HP, tm * 16, kc);       // from LDS
                bf16x16 b = load_fragB_packed(wih, HP, tn * 16, kc);
                acc = wmma_bf16(a, b, acc);
            }
            int m0 = mblk * 128 + tm * 16 + ((lane >> 4) << 3);
#pragma unroll
            for (int r = 0; r < 8; ++r) pre[(size_t)(m0 + r) * G4 + n] = acc[r] + bv;
        }
    }
}

// sequential LSTM scan: one workgroup (32 waves), WMMA step GEMM M=16 N=1200 K=320.
// Per-step 16x1200 f32 `pre` slice is double-buffered in LDS via TDM: at step t the
// tile for t+1 is DMA'd while WMMAs run; s_wait_tensorcnt(1) retires the current tile.
__global__ void k_lstm_seq(const float* pre, const unsigned short* whh, float* mOut){
    extern __shared__ char smem[];
    float*          pb0  = (float*)smem;                     // 16 x 1200
    float*          pb1  = pb0 + 16 * G4;                    // 16 x 1200
    unsigned short* hb   = (unsigned short*)(pb1 + 16 * G4); // 16 x 320 bf16
    float*          cbuf = (float*)(hb + 16 * HP);           // 16 x 300
    float*          hsum = cbuf + 16 * H_;                   // 16
    int tid = threadIdx.x, wid = tid >> 5, lane = tid & 31, nw = blockDim.x >> 5;

    for (int i = tid; i < 16 * HP; i += blockDim.x) hb[i] = 0;
    for (int i = tid; i < 16 * H_; i += blockDim.x) cbuf[i] = 0.f;
    __syncthreads();

    if (wid == 0)   // kick off tile for t=0 (16 rows strided by T_*G4 floats)
        tdm_load_2d((unsigned)(size_t)(void*)pb0, pre,
                    G4, 16, G4, 16, (unsigned long long)T_ * G4, 2);

    for (int t = 0; t < T_; ++t){
        if (tid < 16) hsum[tid] = 0.f;
        float* Gp = (t & 1) ? pb1 : pb0;
        float* Gn = (t & 1) ? pb0 : pb1;
        if (wid == 0){
            if (t + 1 < T_){
                tdm_load_2d((unsigned)(size_t)(void*)Gn, pre + (size_t)(t + 1) * G4,
                            G4, 16, G4, 16, (unsigned long long)T_ * G4, 2);
                __builtin_amdgcn_s_wait_tensorcnt(1);   // in-order: current tile done
            } else {
                __builtin_amdgcn_s_wait_tensorcnt(0);
            }
        }
        __syncthreads();
        // phase 1: Gp = h @ Whh^T + Gp   (gate pre-activations, in place in LDS tile)
        for (int tn = wid; tn < G4 / 16; tn += nw){
            f32x8 acc = {0.f,0.f,0.f,0.f,0.f,0.f,0.f,0.f};
            for (int kc = 0; kc < HP; kc += 32){
                bf16x16 a = load_fragA_row(hb, HP, 0, kc);
                bf16x16 b = load_fragB_packed(whh, HP, tn * 16, kc);
                acc = wmma_bf16(a, b, acc);
            }
            int n  = tn * 16 + (lane & 15);
            int m0 = (lane >> 4) << 3;
#pragma unroll
            for (int r = 0; r < 8; ++r){
                int m = m0 + r;
                Gp[m * G4 + n] += acc[r];
            }
        }
        __syncthreads();
        // phase 2: gate nonlinearities, state update, h-mean accumulation
        for (int u = tid; u < 16 * H_; u += blockDim.x){
            int b = u / H_, j = u % H_;
            float gi = Gp[b * G4 + j];
            float gf = Gp[b * G4 + H_ + j];
            float gg = Gp[b * G4 + 2 * H_ + j];
            float go = Gp[b * G4 + 3 * H_ + j];
            float c = sigmf(gf) * cbuf[u] + sigmf(gi) * tanhf(gg);
            float h = sigmf(go) * tanhf(c);
            cbuf[u] = c;
            hb[b * HP + j] = f2bf(h);
            atomicAdd(&hsum[b], h);
        }
        __syncthreads();
        if (tid < 16) mOut[(size_t)tid * T_ + t] = hsum[tid] * (1.f / H_);
        __syncthreads();
    }
}

__global__ void k_fuse(const float* m1, const float* m2, const float* m3, const float* m4,
                       const float* fw, unsigned short* fb){
    for (int i = blockIdx.x * blockDim.x + threadIdx.x; i < ROWS; i += gridDim.x * blockDim.x){
        float f = fw[0] * m1[i] + fw[1] * m2[i] + fw[2] * m3[i] + fw[3] * m4[i];
        fb[i] = f2bf(f);
    }
}

// h1 = silu(fused @ fc1W^T + b); M=16 N=256 K=4096 (single block, 8 waves)
__global__ void k_fc1(const unsigned short* fb, const unsigned short* w1, const float* b1, float* h1){
    int wv = threadIdx.x >> 5, lane = threadIdx.x & 31;
    for (int tn = wv; tn < 16; tn += 8){
        f32x8 acc = {0.f,0.f,0.f,0.f,0.f,0.f,0.f,0.f};
        for (int kc = 0; kc < T_; kc += 32){
            bf16x16 a = load_fragA_row(fb, T_, 0, kc);
            bf16x16 b = load_fragB_packed(w1, T_, tn * 16, kc);
            acc = wmma_bf16(a, b, acc);
        }
        int n  = tn * 16 + (lane & 15);
        int m0 = (lane >> 4) << 3;
#pragma unroll
        for (int r = 0; r < 8; ++r){
            float z = acc[r] + b1[n];
            h1[(m0 + r) * 256 + n] = z * sigmf(z);
        }
    }
}

// fc2 + softmax + fc3 head; block 256 = 16x16
__global__ void k_head(const float* h1, const float* w2, const float* b2,
                       const float* w3, const float* b3, float* out){
    __shared__ float z[16][16];
    __shared__ float pp[16][16];
    int tid = threadIdx.x, b = tid >> 4, j = tid & 15;
    float s = b2[j];
    for (int k = 0; k < 256; ++k) s += h1[b * 256 + k] * w2[j * 256 + k];
    z[b][j] = s; __syncthreads();
    float mx = z[b][0];
    for (int k = 1; k < 16; ++k) mx = fmaxf(mx, z[b][k]);
    float ex = __expf(z[b][j] - mx);
    pp[b][j] = ex; __syncthreads();
    float den = 0.f;
    for (int k = 0; k < 16; ++k) den += pp[b][k];
    __syncthreads();
    pp[b][j] = (ex / den) * w3[j]; __syncthreads();
    if (j == 0){
        float o = b3[0];
        for (int k = 0; k < 16; ++k) o += pp[b][k];
        out[b] = o;
    }
}

// ================= host =================
extern "C" void kernel_launch(void* const* d_in, const int* in_sizes, int n_in,
                              void* d_out, int out_size, void* d_ws, size_t ws_size,
                              hipStream_t stream){
    (void)in_sizes; (void)n_in; (void)out_size; (void)ws_size;
    const int*   x    = (const int*)d_in[0];
    const float* emb  = (const float*)d_in[1];
    const float* wC[5] = {(const float*)d_in[2], (const float*)d_in[4], (const float*)d_in[6],
                          (const float*)d_in[8], (const float*)d_in[10]};
    const float* bC[5] = {(const float*)d_in[3], (const float*)d_in[5], (const float*)d_in[7],
                          (const float*)d_in[9], (const float*)d_in[11]};
    const float* Wih[3] = {(const float*)d_in[12], (const float*)d_in[16], (const float*)d_in[20]};
    const float* Whh[3] = {(const float*)d_in[13], (const float*)d_in[17], (const float*)d_in[21]};
    const float* bih[3] = {(const float*)d_in[14], (const float*)d_in[18], (const float*)d_in[22]};
    const float* bhh[3] = {(const float*)d_in[15], (const float*)d_in[19], (const float*)d_in[23]};
    const float* fuse = (const float*)d_in[24];
    const float* fc1W = (const float*)d_in[25]; const float* fc1b = (const float*)d_in[26];
    const float* fc2W = (const float*)d_in[27]; const float* fc2b = (const float*)d_in[28];
    const float* fc3W = (const float*)d_in[29]; const float* fc3b = (const float*)d_in[30];
    float* out = (float*)d_out;

    char* base = (char*)d_ws; size_t off = 0;
    auto alloc = [&](size_t bytes)->char* {
        char* r = base + off;
        off = (off + bytes + 255) & ~(size_t)255;
        return r;
    };

    const int ksz[5]     = {2, 4, 3, 6, 5};
    const int cstride[5] = {1, 2, 3, 3, 3};
    const int cpad[5]    = {0, 0, 2, 2, 0};
    const int Lneed[5]   = {2047, 1023, 683, 683, 682};

    unsigned short* xc = (unsigned short*)alloc((size_t)ROWS * EP * 2);
    float* m4 = (float*)alloc((size_t)ROWS * 4);
    unsigned short* wr[5]; float* Y[5];
    for (int i = 0; i < 5; ++i) wr[i] = (unsigned short*)alloc((size_t)EP * ksz[i] * EP * 2);
    for (int i = 0; i < 5; ++i) Y[i]  = (float*)alloc((size_t)B_ * Lneed[i] * EP * 4);
    unsigned short *wihP[3], *whhP[3];
    for (int i = 0; i < 3; ++i) wihP[i] = (unsigned short*)alloc((size_t)G4 * HP * 2);
    for (int i = 0; i < 3; ++i) whhP[i] = (unsigned short*)alloc((size_t)G4 * HP * 2);
    unsigned short* fc1Wp = (unsigned short*)alloc((size_t)256 * T_ * 2);
    float* F = (float*)alloc((size_t)ROWS * FCP * 4);
    float* colsum = (float*)alloc((size_t)FCP * 4);
    unsigned short* Fc = (unsigned short*)alloc((size_t)ROWS * FCP * 2);
    float* G = (float*)alloc((size_t)FCP * FCP * 4);
    float* V = (float*)alloc((size_t)FCP * FCP * 4);
    unsigned short* top = (unsigned short*)alloc((size_t)NPCP * FCP * 2);
    unsigned short* xseq = (unsigned short*)alloc((size_t)ROWS * HP * 2);
    float* pre = (float*)alloc((size_t)ROWS * G4 * 4);
    float* mB[3];
    for (int i = 0; i < 3; ++i) mB[i] = (float*)alloc((size_t)ROWS * 4);
    unsigned short* fusedbf = (unsigned short*)alloc((size_t)ROWS * 2);
    float* h1 = (float*)alloc((size_t)16 * 256 * 4);

    const size_t lstm_smem = (size_t)2 * 16 * G4 * 4 + (size_t)16 * HP * 2 + (size_t)16 * H_ * 4 + 16 * 4;
    const size_t pre_smem  = (size_t)128 * HP * 2;
    const size_t proj_smem = (size_t)128 * FCP * 2;
    (void)hipFuncSetAttribute(reinterpret_cast<const void*>(k_lstm_seq),
                              hipFuncAttributeMaxDynamicSharedMemorySize, (int)lstm_smem);
    (void)hipFuncSetAttribute(reinterpret_cast<const void*>(k_pre_gemm),
                              hipFuncAttributeMaxDynamicSharedMemorySize, (int)pre_smem);
    (void)hipFuncSetAttribute(reinterpret_cast<const void*>(k_proj_gemm),
                              hipFuncAttributeMaxDynamicSharedMemorySize, (int)proj_smem);

    auto gemm_blocks = [](long tiles)->int { return (int)((tiles + 7) / 8); };

    // 1) embedding gather + m4
    k_embed_pack<<<ROWS, 128, 0, stream>>>(x, emb, xc, m4);

    // 2) weight packing
    for (int i = 0; i < 5; ++i)
        k_pack_conv<<<1024, 256, 0, stream>>>(wC[i], wr[i], ksz[i]);
    for (int i = 0; i < 3; ++i){
        k_pack_rows<<<1500, 256, 0, stream>>>(Wih[i], wihP[i], G4, H_, HP);
        k_pack_rows<<<1500, 256, 0, stream>>>(Whh[i], whhP[i], G4, H_, HP);
    }
    k_pack_rows<<<4096, 256, 0, stream>>>(fc1W, fc1Wp, 256, T_, T_);

    // 3) convolutions as WMMA GEMMs
    for (int i = 0; i < 5; ++i){
        long tiles = (long)(EP / 16) * Lneed[i];
        k_conv_gemm<<<gemm_blocks(tiles), 256, 0, stream>>>(xc, wr[i], bC[i], Y[i],
                                                            ksz[i], cstride[i], cpad[i], Lneed[i]);
    }

    // 4) xseq pad init (cols >= NPCP stay zero)
    k_zero_u16<<<8192, 256, 0, stream>>>(xseq, (size_t)ROWS * HP);

    // 5) per-band: assemble -> PCA -> project -> LSTM
    const float* YaB[3] = {Y[0], Y[2], Y[4]};
    const float* YbB[3] = {Y[1], Y[3], Y[4]};
    for (int band = 0; band < 3; ++band){
        k_assemble<<<16384, 256, 0, stream>>>(band, YaB[band], YbB[band], F);
        k_colsum<<<FCP, 256, 0, stream>>>(F, colsum);
        k_center_pack<<<16384, 256, 0, stream>>>(F, colsum, Fc);
        k_cov_gemm<<<gemm_blocks((long)(FCP / 16) * (FCP / 16)), 256, 0, stream>>>(Fc, G);
        k_init_identity<<<1444, 256, 0, stream>>>(V);
        k_jacobi<<<1, 1024, 0, stream>>>(G, V);
        k_eig_pack<<<1, 256, 0, stream>>>(G, V, top);
        k_proj_gemm<<<ROWS / 128, 256, proj_smem, stream>>>(Fc, top, xseq);
        k_pre_gemm<<<ROWS / 128, 256, pre_smem, stream>>>(xseq, wihP[band],
                                                          bih[band], bhh[band], pre);
        k_lstm_seq<<<1, 1024, lstm_smem, stream>>>(pre, whhP[band], mB[band]);
    }

    // 6) fuse + head
    k_fuse<<<256, 256, 0, stream>>>(mB[0], mB[1], mB[2], m4, fuse, fusedbf);
    k_fc1<<<1, 256, 0, stream>>>(fusedbf, fc1Wp, fc1b, h1);
    k_head<<<1, 256, 0, stream>>>(h1, fc2W, fc2b, fc3W, fc3b, out);
}
